// SGMCAILayer_78847009620038
// MI455X (gfx1250) — compile-verified, matrix operand order checked by reading, not run on hardware
//
#include <hip/hip_runtime.h>
#include <hip/hip_bf16.h>
#include <math.h>

typedef __bf16 bf16;
typedef __attribute__((ext_vector_type(16))) __bf16 v16bf;
typedef __attribute__((ext_vector_type(8)))  float  v8f;
typedef __attribute__((ext_vector_type(4)))  int    v4i;

union FragA { v16bf v; uint4 q[2]; bf16 e[16]; };
union Half8 { uint4 q; bf16 e[8]; };

#define BB      2
#define KSEQ    2048
#define DMODEL  1024
#define HEADS   16
#define DHEAD   64
#define DFFN    4096
#define ATTW    256
#define LN_EPS  1e-5f

// ---- CDNA5 async global->LDS staging (guarded; falls back to sync path) ----
#if defined(__has_builtin)
#  if __has_builtin(__builtin_amdgcn_global_load_async_to_lds_b128)
#    define USE_ASYNC_LDS 1
#  endif
#endif

#define GPTR_V4I(p) ((__attribute__((address_space(1))) v4i*)(unsigned long long)(p))
#define LPTR_V4I(p) ((__attribute__((address_space(3))) v4i*)(unsigned int)(unsigned long long)(p))

__device__ inline void wait_async_zero() {
#if defined(__has_builtin)
#  if __has_builtin(__builtin_amdgcn_s_wait_asynccnt)
    __builtin_amdgcn_s_wait_asynccnt(0);
    return;
#  endif
#endif
    asm volatile("s_wait_asynccnt 0" ::: "memory");
}

__device__ inline float half_max16(float x) {
#pragma unroll
    for (int m = 1; m < 16; m <<= 1) x = fmaxf(x, __shfl_xor(x, m, 32));
    return x;
}
__device__ inline float half_sum16(float x) {
#pragma unroll
    for (int m = 1; m < 16; m <<= 1) x += __shfl_xor(x, m, 32);
    return x;
}
__device__ inline v8f v8f_zero() {
    v8f z;
#pragma unroll
    for (int i = 0; i < 8; ++i) z[i] = 0.f;
    return z;
}

// ---------------- fp32 -> bf16 conversion ----------------
__global__ void cvt_bf16_kernel(const float* __restrict__ s, bf16* __restrict__ d, int n) {
    int i = blockIdx.x * blockDim.x + threadIdx.x;
    if (i < n) d[i] = (bf16)s[i];
}

// ---------------- tiled WMMA GEMM: C[M,N] = A[M,K]@B[K,N] + bias ----------------
#define TM 128
#define TN 128
#define TK 32
#define TKP 40   // padded LDS stride (bf16 elems): 80B rows, 16B aligned

__global__ __launch_bounds__(256)
void gemm_bf16_kernel(const bf16* __restrict__ A, const bf16* __restrict__ Bm,
                      const float* __restrict__ bias,
                      float* __restrict__ Cf, bf16* __restrict__ Cb,
                      int M, int N, int Kd, int gelu)
{
    __shared__ __align__(16) bf16 As[TM][TKP];
    __shared__ __align__(16) bf16 Bts[TN][TKP];   // B transposed tile: Bts[n][k]

    const int tid  = threadIdx.x;
    const int lane = tid & 31;
    const int wid  = tid >> 5;
    const int wave_m = wid & 3;    // 4 waves along M (32 rows each)
    const int wave_n = wid >> 2;   // 2 waves along N (64 cols each)
    const int l16 = lane & 15;
    const int hi8 = (lane >> 4) * 8;
    const int mb = blockIdx.y * TM;
    const int nb = blockIdx.x * TN;

    v8f acc[2][4];
#pragma unroll
    for (int mi = 0; mi < 2; ++mi)
#pragma unroll
        for (int ni = 0; ni < 4; ++ni) acc[mi][ni] = v8f_zero();

    const int arow = tid >> 1;
    const int acol = (tid & 1) * 16;

    for (int kt = 0; kt < Kd; kt += TK) {
        // stage A tile (row-major [128][32])
        {
            const bf16* src = A + (size_t)(mb + arow) * Kd + kt + acol;
#if defined(USE_ASYNC_LDS)
            __builtin_amdgcn_global_load_async_to_lds_b128(
                GPTR_V4I(src),     LPTR_V4I(&As[arow][acol]),     0, 0);
            __builtin_amdgcn_global_load_async_to_lds_b128(
                GPTR_V4I(src + 8), LPTR_V4I(&As[arow][acol + 8]), 0, 0);
#else
            uint4 u0 = *(const uint4*)(src);
            uint4 u1 = *(const uint4*)(src + 8);
            *(uint4*)&As[arow][acol]     = u0;
            *(uint4*)&As[arow][acol + 8] = u1;
#endif
        }
        // stage B tile transposed: Bts[n][k] = B[kt+k][nb+n]
#pragma unroll
        for (int c = 0; c < 2; ++c) {
            int chunk = tid * 2 + c;          // 512 chunks of 8
            int krow  = chunk >> 4;           // 0..31
            int n8    = (chunk & 15) * 8;     // 0..120
            Half8 h;
            h.q = *(const uint4*)(Bm + (size_t)(kt + krow) * N + nb + n8);
#pragma unroll
            for (int j = 0; j < 8; ++j) Bts[n8 + j][krow] = h.e[j];
        }
        // prefetch next K tile (emits global_prefetch_b8)
        if (kt + TK < Kd) {
            __builtin_prefetch(A + (size_t)(mb + arow) * Kd + kt + TK + acol, 0, 1);
            __builtin_prefetch(Bm + (size_t)(kt + TK + (tid >> 3)) * N + nb + (tid & 7) * 16, 0, 1);
        }
#if defined(USE_ASYNC_LDS)
        wait_async_zero();
#endif
        __syncthreads();

        FragA a[2], b[4];
#pragma unroll
        for (int mi = 0; mi < 2; ++mi) {
            int m = wave_m * 32 + mi * 16 + l16;
            a[mi].q[0] = *(const uint4*)&As[m][hi8];
            a[mi].q[1] = *(const uint4*)&As[m][hi8 + 16];
        }
#pragma unroll
        for (int ni = 0; ni < 4; ++ni) {
            int n = wave_n * 64 + ni * 16 + l16;
            b[ni].q[0] = *(const uint4*)&Bts[n][hi8];
            b[ni].q[1] = *(const uint4*)&Bts[n][hi8 + 16];
        }
#pragma unroll
        for (int mi = 0; mi < 2; ++mi)
#pragma unroll
            for (int ni = 0; ni < 4; ++ni)
                acc[mi][ni] = __builtin_amdgcn_wmma_f32_16x16x32_bf16(
                    false, a[mi].v, false, b[ni].v, (short)0, acc[mi][ni], false, false);
        __syncthreads();
    }

    // epilogue: bias (+ exact GELU), store f32 and/or bf16
#pragma unroll
    for (int mi = 0; mi < 2; ++mi)
#pragma unroll
        for (int ni = 0; ni < 4; ++ni) {
            int col = nb + wave_n * 64 + ni * 16 + l16;
            float bv = bias ? bias[col] : 0.f;
#pragma unroll
            for (int r = 0; r < 8; ++r) {
                int row = mb + wave_m * 32 + mi * 16 + hi8 + r;
                float vv = acc[mi][ni][r] + bv;
                if (gelu) vv = 0.5f * vv * (1.f + erff(vv * 0.70710678118654752f));
                size_t o = (size_t)row * N + col;
                if (Cf) Cf[o] = vv;
                if (Cb) Cb[o] = (bf16)vv;
            }
        }
}

// ---------------- banded flash attention (WMMA) ----------------
// q,k,v,ctx laid out [B, K, H*DHEAD]; one block = (b, h, 128 rows); 8 waves x 16 rows
__global__ __launch_bounds__(256)
void attn_kernel(const bf16* __restrict__ q, const bf16* __restrict__ kmat,
                 const bf16* __restrict__ vmat, bf16* __restrict__ ctx)
{
    __shared__ __align__(16) bf16 Pl[8][16][32];   // per-wave P staging

    const int lane = threadIdx.x & 31;
    const int wid  = threadIdx.x >> 5;
    const int l16  = lane & 15;
    const int hi8  = (lane >> 4) * 8;

    const int nrb = KSEQ / 128;
    const int rb  = blockIdx.x % nrb;
    const int h   = (blockIdx.x / nrb) % HEADS;
    const int b   = blockIdx.x / (nrb * HEADS);
    const int m0  = rb * 128 + wid * 16;

    const float scale = 0.02209708691207961f;      // 1/sqrt(2048)
    const size_t strideT = (size_t)HEADS * DHEAD;  // elems per time step

    // q A-fragments (16 rows x 64, two k=32 halves)
    FragA aq[2];
    {
        const bf16* qp = q + ((size_t)(b * KSEQ + m0 + l16) * HEADS + h) * DHEAD;
#pragma unroll
        for (int s = 0; s < 2; ++s) {
            aq[s].q[0] = *(const uint4*)(qp + s * 32 + hi8);
            aq[s].q[1] = *(const uint4*)(qp + s * 32 + hi8 + 16);
        }
    }

    v8f o[4];
#pragma unroll
    for (int ni = 0; ni < 4; ++ni) o[ni] = v8f_zero();
    float mrun[8], lrun[8], alpha[8];
#pragma unroll
    for (int r = 0; r < 8; ++r) { mrun[r] = -3.0e38f; lrun[r] = 0.f; }

    int js = m0 - ATTW; if (js < 0) js = 0; js &= ~31;
    int je = m0 + 16 + ATTW; je = (je + 31) & ~31; if (je > KSEQ) je = KSEQ;

    for (int jt = js; jt < je; jt += 32) {
        float p[2][8];
        float tmax[8];
#pragma unroll
        for (int r = 0; r < 8; ++r) tmax[r] = -3.0e38f;

        // scores: two 16-wide tiles, DQ=64 contraction = 2 wmma each
#pragma unroll
        for (int st = 0; st < 2; ++st) {
            int cb = jt + st * 16;
            v8f sc = v8f_zero();
#pragma unroll
            for (int s = 0; s < 2; ++s) {
                FragA bk;
                const bf16* kp = kmat + ((size_t)(b * KSEQ + cb + l16) * HEADS + h) * DHEAD + s * 32;
                bk.q[0] = *(const uint4*)(kp + hi8);
                bk.q[1] = *(const uint4*)(kp + hi8 + 16);
                sc = __builtin_amdgcn_wmma_f32_16x16x32_bf16(
                    false, aq[s].v, false, bk.v, (short)0, sc, false, false);
            }
#pragma unroll
            for (int r = 0; r < 8; ++r) {
                int i = m0 + hi8 + r;
                int j = cb + l16;
                int d = i - j; if (d < 0) d = -d;
                float val = (d <= ATTW) ? sc[r] * scale : -3.0e38f;
                p[st][r] = val;
                tmax[r] = fmaxf(tmax[r], val);
            }
        }

        // online softmax update (per-row stats live replicated in each 16-lane half)
#pragma unroll
        for (int r = 0; r < 8; ++r) {
            float tm   = half_max16(tmax[r]);
            float mnew = fmaxf(mrun[r], tm);
            float a    = (mrun[r] <= -1.0e37f) ? ((mnew <= -1.0e37f) ? 1.f : 0.f)
                                               : __expf(mrun[r] - mnew);
            float rsum = 0.f;
#pragma unroll
            for (int st = 0; st < 2; ++st) {
                float val = p[st][r];
                float pe  = (val <= -1.0e37f) ? 0.f : __expf(val - mnew);
                p[st][r] = pe;
                rsum += pe;
            }
            rsum = half_sum16(rsum);
            lrun[r] = lrun[r] * a + rsum;
            mrun[r] = mnew;
            alpha[r] = a;
        }
#pragma unroll
        for (int ni = 0; ni < 4; ++ni)
#pragma unroll
            for (int r = 0; r < 8; ++r) o[ni][r] *= alpha[r];

        // stage P (C layout -> LDS row-major) then reload as A fragment
#pragma unroll
        for (int st = 0; st < 2; ++st)
#pragma unroll
            for (int r = 0; r < 8; ++r)
                Pl[wid][hi8 + r][st * 16 + l16] = (bf16)p[st][r];

        FragA pa;
        pa.q[0] = *(const uint4*)&Pl[wid][l16][hi8];
        pa.q[1] = *(const uint4*)&Pl[wid][l16][hi8 + 16];

        // ctx += P(16x32) @ V(32x64)
        const bf16* vbase = vmat + ((size_t)(b * KSEQ + jt) * HEADS + h) * DHEAD;
#pragma unroll
        for (int ni = 0; ni < 4; ++ni) {
            FragA bv;
            const bf16* vp = vbase + ni * 16 + l16;
#pragma unroll
            for (int e = 0; e < 8; ++e) {
                bv.e[e]     = vp[(size_t)(hi8 + e) * strideT];
                bv.e[8 + e] = vp[(size_t)(hi8 + 16 + e) * strideT];
            }
            o[ni] = __builtin_amdgcn_wmma_f32_16x16x32_bf16(
                false, pa.v, false, bv.v, (short)0, o[ni], false, false);
        }
    }

    // normalize and write ctx (bf16)
#pragma unroll
    for (int ni = 0; ni < 4; ++ni)
#pragma unroll
        for (int r = 0; r < 8; ++r) {
            int i = m0 + hi8 + r;
            float val = o[ni][r] / lrun[r];
            ctx[((size_t)(b * KSEQ + i) * HEADS + h) * DHEAD + ni * 16 + l16] = (bf16)val;
        }
}

// ---------------- residual add + LayerNorm (D=1024, one block per row) ----------------
__global__ __launch_bounds__(256)
void add_ln_kernel(const float* __restrict__ a, const float* __restrict__ res,
                   const float* __restrict__ g, const float* __restrict__ bb,
                   float* __restrict__ outf, bf16* __restrict__ outb)
{
    __shared__ float red[8];
    const int tid  = threadIdx.x;
    const int lane = tid & 31;
    const int wid  = tid >> 5;
    const size_t row = blockIdx.x;
    const float* pa = a + row * DMODEL;
    const float* pr = res + row * DMODEL;

    float x[4];
    float s = 0.f;
#pragma unroll
    for (int j = 0; j < 4; ++j) {
        int c = tid + j * 256;
        x[j] = pa[c] + pr[c];
        s += x[j];
    }
#pragma unroll
    for (int m = 1; m < 32; m <<= 1) s += __shfl_xor(s, m, 32);
    if (lane == 0) red[wid] = s;
    __syncthreads();
    float tot = 0.f;
#pragma unroll
    for (int w = 0; w < 8; ++w) tot += red[w];
    float mean = tot * (1.0f / DMODEL);
    __syncthreads();

    float vs = 0.f;
#pragma unroll
    for (int j = 0; j < 4; ++j) { float d = x[j] - mean; vs += d * d; }
#pragma unroll
    for (int m = 1; m < 32; m <<= 1) vs += __shfl_xor(vs, m, 32);
    if (lane == 0) red[wid] = vs;
    __syncthreads();
    float vtot = 0.f;
#pragma unroll
    for (int w = 0; w < 8; ++w) vtot += red[w];
    float rstd = rsqrtf(vtot * (1.0f / DMODEL) + LN_EPS);

#pragma unroll
    for (int j = 0; j < 4; ++j) {
        int c = tid + j * 256;
        float y = (x[j] - mean) * rstd * g[c] + bb[c];
        outf[row * DMODEL + c] = y;
        if (outb) outb[row * DMODEL + c] = (bf16)y;
    }
}

// ---------------- launcher ----------------
extern "C" void kernel_launch(void* const* d_in, const int* in_sizes, int n_in,
                              void* d_out, int out_size, void* d_ws, size_t ws_size,
                              hipStream_t stream) {
    (void)in_sizes; (void)n_in; (void)out_size; (void)ws_size;
    const float* x    = (const float*)d_in[0];
    const float* Wq   = (const float*)d_in[1];
    const float* bq   = (const float*)d_in[2];
    const float* Wk   = (const float*)d_in[3];
    const float* bk   = (const float*)d_in[4];
    const float* Wv   = (const float*)d_in[5];
    const float* bv   = (const float*)d_in[6];
    const float* Wo   = (const float*)d_in[7];
    const float* bo   = (const float*)d_in[8];
    const float* W1   = (const float*)d_in[9];
    const float* b1   = (const float*)d_in[10];
    const float* W2   = (const float*)d_in[11];
    const float* b2   = (const float*)d_in[12];
    const float* ln1g = (const float*)d_in[13];
    const float* ln1b = (const float*)d_in[14];
    const float* ln2g = (const float*)d_in[15];
    const float* ln2b = (const float*)d_in[16];

    const size_t M   = (size_t)BB * KSEQ;            // 4096 rows
    const size_t nX  = M * DMODEL;                   // 4M
    const size_t nW  = (size_t)DMODEL * DMODEL;      // 1M
    const size_t nW1 = (size_t)DMODEL * DFFN;        // 4M
    const size_t nH  = M * DFFN;                     // 16M

    char* w = (char*)d_ws;
    size_t off = 0;
    auto take = [&](size_t bytes) { char* p = w + off; off += (bytes + 255) & ~(size_t)255; return p; };

    bf16* xb   = (bf16*)take(nX * 2);
    bf16* Wqb  = (bf16*)take(nW * 2);
    bf16* Wkb  = (bf16*)take(nW * 2);
    bf16* Wvb  = (bf16*)take(nW * 2);
    bf16* Wob  = (bf16*)take(nW * 2);
    bf16* W1b  = (bf16*)take(nW1 * 2);
    bf16* W2b  = (bf16*)take(nW1 * 2);
    bf16* qb   = (bf16*)take(nX * 2);
    bf16* kb   = (bf16*)take(nX * 2);
    bf16* vb   = (bf16*)take(nX * 2);
    bf16* ctxb = (bf16*)take(nX * 2);
    float* sa  = (float*)take(nX * 4);
    float* x1f = (float*)take(nX * 4);
    bf16* x1b  = (bf16*)take(nX * 2);
    bf16* hb   = (bf16*)take(nH * 2);
    float* ff  = (float*)take(nX * 4);

    auto cvt = [&](const float* s, bf16* d, size_t n) {
        cvt_bf16_kernel<<<(unsigned)((n + 255) / 256), 256, 0, stream>>>(s, d, (int)n);
    };
    cvt(x,  xb,  nX);
    cvt(Wq, Wqb, nW);
    cvt(Wk, Wkb, nW);
    cvt(Wv, Wvb, nW);
    cvt(Wo, Wob, nW);
    cvt(W1, W1b, nW1);
    cvt(W2, W2b, nW1);

    // QKV projections
    dim3 gproj(DMODEL / TN, (unsigned)(M / TM));
    gemm_bf16_kernel<<<gproj, 256, 0, stream>>>(xb, Wqb, bq, nullptr, qb, (int)M, DMODEL, DMODEL, 0);
    gemm_bf16_kernel<<<gproj, 256, 0, stream>>>(xb, Wkb, bk, nullptr, kb, (int)M, DMODEL, DMODEL, 0);
    gemm_bf16_kernel<<<gproj, 256, 0, stream>>>(xb, Wvb, bv, nullptr, vb, (int)M, DMODEL, DMODEL, 0);

    // banded attention
    attn_kernel<<<BB * HEADS * (KSEQ / 128), 256, 0, stream>>>(qb, kb, vb, ctxb);

    // output projection + residual LN1
    gemm_bf16_kernel<<<gproj, 256, 0, stream>>>(ctxb, Wob, bo, sa, nullptr, (int)M, DMODEL, DMODEL, 0);
    add_ln_kernel<<<(unsigned)M, 256, 0, stream>>>(sa, x, ln1g, ln1b, x1f, x1b);

    // FFN
    dim3 gff1(DFFN / TN, (unsigned)(M / TM));
    gemm_bf16_kernel<<<gff1, 256, 0, stream>>>(x1b, W1b, b1, nullptr, hb, (int)M, DFFN, DMODEL, 1);
    gemm_bf16_kernel<<<gproj, 256, 0, stream>>>(hb, W2b, b2, ff, nullptr, (int)M, DMODEL, DFFN, 0);
    add_ln_kernel<<<(unsigned)M, 256, 0, stream>>>(ff, x1f, ln2g, ln2b, (float*)d_out, nullptr);
}
